// BillboardAllocatorGNN_1623497638440
// MI455X (gfx1250) — compile-verified, matrix operand order checked by reading, not run on hardware
//
#include <hip/hip_runtime.h>
#include <hip/hip_bf16.h>

// ---------------- problem constants ----------------
constexpr int B  = 4;
constexpr int N  = 20000;
constexpr int E  = 320000;
constexpr int FN = 16;     // F_NODE
constexpr int FA = 8;      // F_AD
constexpr int H  = 64;
constexpr int L  = 2;
constexpr int HEADS = 4;
constexpr int DH = H / HEADS;   // 16

typedef _Float16 half16 __attribute__((ext_vector_type(16)));
typedef float    float8 __attribute__((ext_vector_type(8)));

// Async 16B global->LDS copy, SADDR form. ISA: IOFFSET is added to BOTH the
// LDS address (VDST VGPR) and the global address (SADDR + VADDR voffset),
// so consecutive chunks need only a literal offset. Tracked on ASYNCcnt.
#define ASYNC_LD16(ldsa, voff, sbase, OFS)                               \
  asm volatile("global_load_async_to_lds_b128 %0, %1, %2 offset:" OFS    \
               :: "v"(ldsa), "v"(voff), "s"(sbase) : "memory")
#define ASYNC_WAIT() asm volatile("s_wait_asynccnt 0x0" ::: "memory")

// ---------------- workspace layout (bytes, 256-aligned) ----------------
constexpr size_t OFF_H16  = 0;                         // B*N*64 f16   = 10,240,000
constexpr size_t OFF_AGG  = 10240000;                  // B*N*64 f32   = 20,480,000
constexpr size_t OFF_K16  = 30720000;                  // B*N*64 f16
constexpr size_t OFF_V16  = 40960000;                  // B*N*64 f16
constexpr size_t OFF_WTN  = 51200000;                  // w_node^T f16 (64x16)
constexpr size_t OFF_WMSG = 51204096;                  // msg_w^T  f16 (L x 64 x 128)
constexpr size_t OFF_WUPD = 51236864;                  // upd_w^T  f16 (L x 64 x 128)
constexpr size_t OFF_WK   = 51269632;                  // wk^T f16 (64x64)
constexpr size_t OFF_WV   = 51277824;                  // wv^T f16 (64x64)
constexpr size_t OFF_A    = 51286016;                  // a   f32 (B x 64)
constexpr size_t OFF_Q    = 51287040;                  // q   f32 (B x 64)
constexpr size_t OFF_CTX  = 51288064;                  // ctx f32 (B x 64)
constexpr size_t OFF_G    = 51289088;                  // g   f32 (B x 64)
constexpr size_t SZ_AGG   = (size_t)B * N * 64 * 4;

// =====================================================================
// 0) Transpose + f16-convert weights (N-major, K-minor) for WMMA B-frags
// =====================================================================
__global__ void prep_weights(const float* __restrict__ w_node,
                             const float* __restrict__ msg_w,
                             const float* __restrict__ upd_w,
                             const float* __restrict__ wk,
                             const float* __restrict__ wv,
                             _Float16* __restrict__ wnode_t,
                             _Float16* __restrict__ msgw_t,
                             _Float16* __restrict__ updw_t,
                             _Float16* __restrict__ wk_t,
                             _Float16* __restrict__ wv_t) {
  int tid = blockIdx.x * blockDim.x + threadIdx.x;
  int nth = gridDim.x * blockDim.x;
  for (int i = tid; i < 64 * 16; i += nth) {           // (16,64) -> (64,16)
    int n = i / 16, k = i % 16;
    wnode_t[i] = (_Float16)w_node[k * 64 + n];
  }
  for (int i = tid; i < L * 64 * 128; i += nth) {      // (L,128,64) -> (L,64,128)
    int l = i / (64 * 128);
    int r = i % (64 * 128);
    int n = r / 128, k = r % 128;
    msgw_t[i] = (_Float16)msg_w[(l * 128 + k) * 64 + n];
    updw_t[i] = (_Float16)upd_w[(l * 128 + k) * 64 + n];
  }
  for (int i = tid; i < 64 * 64; i += nth) {           // (64,64) -> (64,64)^T
    int n = i / 64, k = i % 64;
    wk_t[i] = (_Float16)wk[k * 64 + n];
    wv_t[i] = (_Float16)wv[k * 64 + n];
  }
}

// =====================================================================
// 1) h = relu(nodes @ w_node + b)  -> f16
// =====================================================================
__global__ void node_embed(const float* __restrict__ nodes,
                           const float* __restrict__ w,
                           const float* __restrict__ bias,
                           _Float16* __restrict__ h16) {
  __shared__ float sw[FN * H];
  __shared__ float sb[H];
  for (int i = threadIdx.x; i < FN * H; i += blockDim.x) sw[i] = w[i];
  for (int i = threadIdx.x; i < H; i += blockDim.x)      sb[i] = bias[i];
  __syncthreads();
  long idx = (long)blockIdx.x * blockDim.x + threadIdx.x;
  if (idx >= (long)B * N) return;
  float x[FN];
#pragma unroll
  for (int i = 0; i < FN; i++) x[i] = nodes[idx * FN + i];
  _Float16* out = h16 + idx * H;
  for (int j = 0; j < H; j++) {
    float acc = sb[j];
#pragma unroll
    for (int i = 0; i < FN; i++) acc += x[i] * sw[i * H + j];
    out[j] = (_Float16)(acc > 0.f ? acc : 0.f);
  }
}

// =====================================================================
// 2) Edge MLP + scatter:  msg = relu([h_src||h_dst] @ msg_w + b)
//    agg[dst] += msg.   One wave per 16-edge tile, WMMA 16x16x32 f16.
//    All LDS staging via GLOBAL_LOAD_ASYNC_TO_LDS (ASYNCcnt, no VGPR data).
// =====================================================================
__global__ void edge_msg(const _Float16* __restrict__ h16,
                         const int* __restrict__ links,
                         const _Float16* __restrict__ msgw_t,
                         const float* __restrict__ msg_b,
                         float* __restrict__ agg, int layer) {
  const int b    = blockIdx.y;
  const int w    = threadIdx.x >> 5;
  const int lane = threadIdx.x & 31;
  __shared__ __align__(32) _Float16 s_w[64 * 128];       // weights (N x K), 16 KB
  __shared__ __align__(32) _Float16 s_a[8][16 * 128];    // per-wave A tile
  __shared__ int s_dst[8][16];

  { // async weight stage: thread t copies chunks t, t+256, t+512, t+768
    unsigned ldsa = (unsigned)(size_t)(&s_w[0]) + (unsigned)(threadIdx.x * 16);
    unsigned voff = (unsigned)(layer * 16384 + threadIdx.x * 16);
    ASYNC_LD16(ldsa, voff, msgw_t, "0");
    ASYNC_LD16(ldsa, voff, msgw_t, "4096");
    ASYNC_LD16(ldsa, voff, msgw_t, "8192");
    ASYNC_LD16(ldsa, voff, msgw_t, "12288");
  }

  const int  tile = blockIdx.x * 8 + w;
  const long eb   = (long)tile * 16;
  const int  m    = lane & 15;          // tile row this lane gathers
  const int  part = lane >> 4;          // 0 = src half, 1 = dst half
  const int* lk   = links + (long)b * 2 * E + (long)part * E;
  const int  idx  = lk[eb + m];
  __builtin_prefetch(lk + eb + m + 128, 0, 1);           // -> global_prefetch_b8
  if (part == 1) s_dst[w][m] = idx;
  { // async gather of one 64-half row (128 B) into A tile
    unsigned ldsa = (unsigned)(size_t)(&s_a[w][m * 128 + part * 64]);
    unsigned voff = ((unsigned)(b * N + idx)) * 128u;    // byte offset < 2^32
    ASYNC_LD16(ldsa, voff, h16, "0");
    ASYNC_LD16(ldsa, voff, h16, "16");
    ASYNC_LD16(ldsa, voff, h16, "32");
    ASYNC_LD16(ldsa, voff, h16, "48");
    ASYNC_LD16(ldsa, voff, h16, "64");
    ASYNC_LD16(ldsa, voff, h16, "80");
    ASYNC_LD16(ldsa, voff, h16, "96");
    ASYNC_LD16(ldsa, voff, h16, "112");
  }
  ASYNC_WAIT();                                          // drain own asyncs
  __syncthreads();                                       // publish LDS

  const int row  = lane & 15;
  const int koff = (lane >> 4) * 16;
  half16 afrag[4];
#pragma unroll
  for (int ks = 0; ks < 4; ks++)
    afrag[ks] = *(const half16*)(&s_a[w][row * 128 + ks * 32 + koff]);

  const int n  = lane & 15;
  const int mb = (lane >> 4) * 8;
  // 32-bit zero-extended scatter offsets -> saddr-form atomics
  unsigned ofs[8];
#pragma unroll
  for (int r = 0; r < 8; r++)
    ofs[r] = ((unsigned)(b * N + s_dst[w][mb + r])) * 64u + (unsigned)n;
  const float* bias = msg_b + layer * 64;
  float bvv[4];
#pragma unroll
  for (int nt = 0; nt < 4; nt++) bvv[nt] = bias[nt * 16 + n];

#pragma unroll
  for (int nt = 0; nt < 4; nt++) {
    float8 acc = {0.f, 0.f, 0.f, 0.f, 0.f, 0.f, 0.f, 0.f};
#pragma unroll
    for (int ks = 0; ks < 4; ks++) {
      half16 bfrag = *(const half16*)(&s_w[(nt * 16 + row) * 128 + ks * 32 + koff]);
      acc = __builtin_amdgcn_wmma_f32_16x16x32_f16(false, afrag[ks], false, bfrag,
                                                   (short)0, acc, false, false);
    }
#pragma unroll
    for (int r = 0; r < 8; r++) {
      float v = acc[r] + bvv[nt];
      v = v > 0.f ? v : 0.f;
      atomicAdd(&agg[(size_t)ofs[r] + (size_t)(nt * 16)], v);
    }
  }
}

// =====================================================================
// 3) Node update: h += relu([h||agg] @ upd_w + b)    (WMMA, in-place f16)
// =====================================================================
__global__ void node_update(_Float16* __restrict__ h16,
                            const float* __restrict__ agg,
                            const _Float16* __restrict__ updw_t,
                            const float* __restrict__ upd_b, int layer) {
  const int b    = blockIdx.y;
  const int w    = threadIdx.x >> 5;
  const int lane = threadIdx.x & 31;
  __shared__ __align__(32) _Float16 s_w[64 * 128];
  __shared__ __align__(32) _Float16 s_a[8][16 * 128];

  { // async weight stage
    unsigned ldsa = (unsigned)(size_t)(&s_w[0]) + (unsigned)(threadIdx.x * 16);
    unsigned voff = (unsigned)(layer * 16384 + threadIdx.x * 16);
    ASYNC_LD16(ldsa, voff, updw_t, "0");
    ASYNC_LD16(ldsa, voff, updw_t, "4096");
    ASYNC_LD16(ldsa, voff, updw_t, "8192");
    ASYNC_LD16(ldsa, voff, updw_t, "12288");
  }

  const int tile  = blockIdx.x * 8 + w;
  const int node0 = tile * 16;
  const bool active = node0 < N;

  if (active) {
    const int m = lane & 15, part = lane >> 4;
    const int node = node0 + m;
    if (part == 0) {                                     // h -> cols [0,64), async
      unsigned ldsa = (unsigned)(size_t)(&s_a[w][m * 128]);
      unsigned voff = ((unsigned)(b * N + node)) * 128u;
      ASYNC_LD16(ldsa, voff, h16, "0");
      ASYNC_LD16(ldsa, voff, h16, "16");
      ASYNC_LD16(ldsa, voff, h16, "32");
      ASYNC_LD16(ldsa, voff, h16, "48");
      ASYNC_LD16(ldsa, voff, h16, "64");
      ASYNC_LD16(ldsa, voff, h16, "80");
      ASYNC_LD16(ldsa, voff, h16, "96");
      ASYNC_LD16(ldsa, voff, h16, "112");
    } else {                                             // agg f32 -> f16 cols [64,128)
      const float* ar = agg + (size_t)((unsigned)(b * N + node)) * 64u;
#pragma unroll 8
      for (int i = 0; i < 64; i++) s_a[w][m * 128 + 64 + i] = (_Float16)ar[i];
    }
  }
  ASYNC_WAIT();                                          // all waves drain
  __syncthreads();

  if (active) {
    const int row  = lane & 15;
    const int koff = (lane >> 4) * 16;
    half16 afrag[4];
#pragma unroll
    for (int ks = 0; ks < 4; ks++)
      afrag[ks] = *(const half16*)(&s_a[w][row * 128 + ks * 32 + koff]);
    const int n  = lane & 15;
    const int mb = (lane >> 4) * 8;
    unsigned hofs[8];
#pragma unroll
    for (int r = 0; r < 8; r++)
      hofs[r] = ((unsigned)(b * N + node0 + mb + r)) * 64u + (unsigned)n;
    const float* bias = upd_b + layer * 64;
    float bvv[4];
#pragma unroll
    for (int nt = 0; nt < 4; nt++) bvv[nt] = bias[nt * 16 + n];

#pragma unroll
    for (int nt = 0; nt < 4; nt++) {
      float8 acc = {0.f, 0.f, 0.f, 0.f, 0.f, 0.f, 0.f, 0.f};
#pragma unroll
      for (int ks = 0; ks < 4; ks++) {
        half16 bfrag = *(const half16*)(&s_w[(nt * 16 + row) * 128 + ks * 32 + koff]);
        acc = __builtin_amdgcn_wmma_f32_16x16x32_f16(false, afrag[ks], false, bfrag,
                                                     (short)0, acc, false, false);
      }
#pragma unroll
      for (int r = 0; r < 8; r++) {
        float v = acc[r] + bvv[nt];
        v = v > 0.f ? v : 0.f;
        float hold = (float)s_a[w][(mb + r) * 128 + nt * 16 + n];
        h16[(size_t)hofs[r] + (size_t)(nt * 16)] = (_Float16)(hold + v);
      }
    }
  }
}

// =====================================================================
// 4) k/v projections: k = h@wk+bk, v = h@wv+bv   (WMMA, K=64)
// =====================================================================
__global__ void kv_proj(const _Float16* __restrict__ h16,
                        const _Float16* __restrict__ wk_t,
                        const _Float16* __restrict__ wv_t,
                        const float* __restrict__ bk,
                        const float* __restrict__ bvec,
                        _Float16* __restrict__ k16,
                        _Float16* __restrict__ v16) {
  const int b    = blockIdx.y;
  const int w    = threadIdx.x >> 5;
  const int lane = threadIdx.x & 31;
  __shared__ __align__(32) _Float16 s_wk[64 * 64];
  __shared__ __align__(32) _Float16 s_wv[64 * 64];
  __shared__ __align__(32) _Float16 s_a[8][16 * 64];

  { // async weight stage: 2 x 8KB
    unsigned lk_a = (unsigned)(size_t)(&s_wk[0]) + (unsigned)(threadIdx.x * 16);
    unsigned lv_a = (unsigned)(size_t)(&s_wv[0]) + (unsigned)(threadIdx.x * 16);
    unsigned voff = (unsigned)(threadIdx.x * 16);
    ASYNC_LD16(lk_a, voff, wk_t, "0");
    ASYNC_LD16(lk_a, voff, wk_t, "4096");
    ASYNC_LD16(lv_a, voff, wv_t, "0");
    ASYNC_LD16(lv_a, voff, wv_t, "4096");
  }

  const int tile  = blockIdx.x * 8 + w;
  const int node0 = tile * 16;
  const bool active = node0 < N;

  if (active) { // 2 lanes per row: each copies 32 halfs (64 B), async
    const int m = lane >> 1, hs = lane & 1;
    unsigned ldsa = (unsigned)(size_t)(&s_a[w][m * 64 + hs * 32]);
    unsigned voff = ((unsigned)(b * N + node0 + m)) * 128u + (unsigned)(hs * 64);
    ASYNC_LD16(ldsa, voff, h16, "0");
    ASYNC_LD16(ldsa, voff, h16, "16");
    ASYNC_LD16(ldsa, voff, h16, "32");
    ASYNC_LD16(ldsa, voff, h16, "48");
  }
  ASYNC_WAIT();
  __syncthreads();

  if (active) {
    const int row  = lane & 15;
    const int koff = (lane >> 4) * 16;
    half16 afrag[2];
#pragma unroll
    for (int ks = 0; ks < 2; ks++)
      afrag[ks] = *(const half16*)(&s_a[w][row * 64 + ks * 32 + koff]);
    const int n  = lane & 15;
    const int mb = (lane >> 4) * 8;
    unsigned ofs[8];
#pragma unroll
    for (int r = 0; r < 8; r++)
      ofs[r] = ((unsigned)(b * N + node0 + mb + r)) * 64u + (unsigned)n;
    float bkv[4], bvv[4];
#pragma unroll
    for (int nt = 0; nt < 4; nt++) { bkv[nt] = bk[nt * 16 + n]; bvv[nt] = bvec[nt * 16 + n]; }

#pragma unroll
    for (int nt = 0; nt < 4; nt++) {
      float8 ak = {0.f, 0.f, 0.f, 0.f, 0.f, 0.f, 0.f, 0.f};
      float8 av = {0.f, 0.f, 0.f, 0.f, 0.f, 0.f, 0.f, 0.f};
#pragma unroll
      for (int ks = 0; ks < 2; ks++) {
        half16 bk_f = *(const half16*)(&s_wk[(nt * 16 + row) * 64 + ks * 32 + koff]);
        half16 bv_f = *(const half16*)(&s_wv[(nt * 16 + row) * 64 + ks * 32 + koff]);
        ak = __builtin_amdgcn_wmma_f32_16x16x32_f16(false, afrag[ks], false, bk_f,
                                                    (short)0, ak, false, false);
        av = __builtin_amdgcn_wmma_f32_16x16x32_f16(false, afrag[ks], false, bv_f,
                                                    (short)0, av, false, false);
      }
#pragma unroll
      for (int r = 0; r < 8; r++) {
        k16[(size_t)ofs[r] + (size_t)(nt * 16)] = (_Float16)(ak[r] + bkv[nt]);
        v16[(size_t)ofs[r] + (size_t)(nt * 16)] = (_Float16)(av[r] + bvv[nt]);
      }
    }
  }
}

// =====================================================================
// 5) a = relu(ad@w_ad + b_ad); q = a@wq + bq   (tiny; 1 block)
// =====================================================================
__global__ void aq_kernel(const float* __restrict__ ad,
                          const float* __restrict__ w_ad,
                          const float* __restrict__ b_ad,
                          const float* __restrict__ wq,
                          const float* __restrict__ bq,
                          float* __restrict__ a_out,
                          float* __restrict__ q_out) {
  __shared__ float s_a[B][H];
  int b = threadIdx.x / H, j = threadIdx.x % H;  // 256 = B*H exactly
  float acc = b_ad[j];
#pragma unroll
  for (int i = 0; i < FA; i++) acc += ad[b * FA + i] * w_ad[i * H + j];
  float av = acc > 0.f ? acc : 0.f;
  s_a[b][j] = av;
  a_out[b * H + j] = av;
  __syncthreads();
  float qv = bq[j];
  for (int i = 0; i < H; i++) qv += s_a[b][i] * wq[i * H + j];
  q_out[b * H + j] = qv;
}

// =====================================================================
// 6) attention: softmax(q.k/4) . v over N, per (b,head); online softmax
// =====================================================================
__global__ void attn_kernel(const _Float16* __restrict__ k16,
                            const _Float16* __restrict__ v16,
                            const float* __restrict__ q,
                            float* __restrict__ ctx) {
  const int b  = blockIdx.x >> 2;
  const int hd = blockIdx.x & 3;
  const int tid = threadIdx.x;
  float qv[DH];
#pragma unroll
  for (int i = 0; i < DH; i++) qv[i] = q[b * H + hd * DH + i];
  float m = -1e30f, l = 0.f, acc[DH];
#pragma unroll
  for (int i = 0; i < DH; i++) acc[i] = 0.f;
  for (int n = tid; n < N; n += 256) {
    const _Float16* kr = k16 + ((long)b * N + n) * H + hd * DH;
    float s = 0.f;
#pragma unroll
    for (int i = 0; i < DH; i++) s += qv[i] * (float)kr[i];
    s *= 0.25f;                                   // 1/sqrt(16)
    float nm = fmaxf(m, s);
    float sc = __expf(m - nm);
    float p  = __expf(s - nm);
    l = l * sc + p;
    const _Float16* vr = v16 + ((long)b * N + n) * H + hd * DH;
#pragma unroll
    for (int i = 0; i < DH; i++) acc[i] = acc[i] * sc + p * (float)vr[i];
    m = nm;
  }
  __shared__ float sm[256], sl[256], sv[DH][256];
  sm[tid] = m; sl[tid] = l;
#pragma unroll
  for (int i = 0; i < DH; i++) sv[i][tid] = acc[i];
  __syncthreads();
  for (int off = 128; off > 0; off >>= 1) {
    if (tid < off) {
      float m2 = sm[tid + off], l2 = sl[tid + off];
      float nm = fmaxf(sm[tid], m2);
      float s1 = __expf(sm[tid] - nm), s2 = __expf(m2 - nm);
      sl[tid] = sl[tid] * s1 + l2 * s2;
#pragma unroll
      for (int i = 0; i < DH; i++) sv[i][tid] = sv[i][tid] * s1 + sv[i][tid + off] * s2;
      sm[tid] = nm;
    }
    __syncthreads();
  }
  if (tid < DH) ctx[b * H + hd * DH + tid] = sv[tid][0] / sl[0];
}

// =====================================================================
// 7) g = layernorm(a + ctx@wo + bo) * ln_g + ln_b    (1 block, 256 thr)
// =====================================================================
__global__ void g_kernel(const float* __restrict__ a,
                         const float* __restrict__ ctx,
                         const float* __restrict__ wo,
                         const float* __restrict__ bo,
                         const float* __restrict__ ln_g,
                         const float* __restrict__ ln_b,
                         float* __restrict__ g) {
  __shared__ float st[B][H];
  int b = threadIdx.x / H, j = threadIdx.x % H;
  float t = bo[j] + a[b * H + j];
  for (int i = 0; i < H; i++) t += ctx[b * H + i] * wo[i * H + j];
  st[b][j] = t;
  __syncthreads();
  float mu = 0.f;
  for (int i = 0; i < H; i++) mu += st[b][i];
  mu *= (1.f / H);
  float var = 0.f;
  for (int i = 0; i < H; i++) { float d = st[b][i] - mu; var += d * d; }
  var *= (1.f / H);
  g[b * H + j] = (t - mu) * rsqrtf(var + 1e-5f) * ln_g[j] + ln_b[j];
}

// =====================================================================
// 8) logits[b,n] = h.g/8 + h.policy_w + policy_b, masked
// =====================================================================
__global__ void logits_kernel(const _Float16* __restrict__ h16,
                              const float* __restrict__ g,
                              const float* __restrict__ pw,
                              const float* __restrict__ pb,
                              const unsigned char* __restrict__ mask,
                              float* __restrict__ out) {
  long i = (long)blockIdx.x * blockDim.x + threadIdx.x;
  if (i >= (long)B * N) return;
  int b = (int)(i / N);
  const _Float16* hr = h16 + i * 64;
  __builtin_prefetch(hr + 64 * 256, 0, 1);
  float dg = 0.f, dp = 0.f;
#pragma unroll 8
  for (int j = 0; j < H; j++) {
    float hv = (float)hr[j];
    dg += hv * g[b * H + j];
    dp += hv * pw[j];
  }
  float logit = dg * 0.125f + dp + pb[0];
  out[i] = mask[i] ? logit : -1e9f;
}

// =====================================================================
extern "C" void kernel_launch(void* const* d_in, const int* in_sizes, int n_in,
                              void* d_out, int out_size, void* d_ws, size_t ws_size,
                              hipStream_t stream) {
  (void)in_sizes; (void)n_in; (void)out_size; (void)ws_size;
  const float* graph_nodes = (const float*)d_in[0];
  const float* current_ad  = (const float*)d_in[1];
  const float* w_node      = (const float*)d_in[2];
  const float* b_node      = (const float*)d_in[3];
  const float* msg_w       = (const float*)d_in[4];
  const float* msg_b       = (const float*)d_in[5];
  const float* upd_w       = (const float*)d_in[6];
  const float* upd_b       = (const float*)d_in[7];
  const float* w_ad        = (const float*)d_in[8];
  const float* b_ad        = (const float*)d_in[9];
  const float* wq          = (const float*)d_in[10];
  const float* bq          = (const float*)d_in[11];
  const float* wk          = (const float*)d_in[12];
  const float* bk          = (const float*)d_in[13];
  const float* wv          = (const float*)d_in[14];
  const float* bv          = (const float*)d_in[15];
  const float* wo          = (const float*)d_in[16];
  const float* bo          = (const float*)d_in[17];
  const float* ln_g        = (const float*)d_in[18];
  const float* ln_b        = (const float*)d_in[19];
  const float* policy_w    = (const float*)d_in[20];
  const float* policy_b    = (const float*)d_in[21];
  const int*   edge_links  = (const int*)d_in[22];
  const unsigned char* mask = (const unsigned char*)d_in[23];

  char* ws = (char*)d_ws;
  _Float16* h16    = (_Float16*)(ws + OFF_H16);
  float*    agg    = (float*)   (ws + OFF_AGG);
  _Float16* k16    = (_Float16*)(ws + OFF_K16);
  _Float16* v16    = (_Float16*)(ws + OFF_V16);
  _Float16* wnodeT = (_Float16*)(ws + OFF_WTN);
  _Float16* wmsgT  = (_Float16*)(ws + OFF_WMSG);
  _Float16* wupdT  = (_Float16*)(ws + OFF_WUPD);
  _Float16* wkT    = (_Float16*)(ws + OFF_WK);
  _Float16* wvT    = (_Float16*)(ws + OFF_WV);
  float*    a_buf  = (float*)(ws + OFF_A);
  float*    q_buf  = (float*)(ws + OFF_Q);
  float*    ctx    = (float*)(ws + OFF_CTX);
  float*    g_buf  = (float*)(ws + OFF_G);

  prep_weights<<<64, 256, 0, stream>>>(w_node, msg_w, upd_w, wk, wv,
                                       wnodeT, wmsgT, wupdT, wkT, wvT);
  (void)wnodeT;

  node_embed<<<(B * N + 255) / 256, 256, 0, stream>>>(graph_nodes, w_node, b_node, h16);

  const int edge_blocks = E / (16 * 8);            // 2500, exact
  const int node_blocks = (N / 16 + 7) / 8;        // 157
  for (int l = 0; l < L; l++) {
    hipMemsetAsync(agg, 0, SZ_AGG, stream);
    edge_msg<<<dim3(edge_blocks, B), 256, 0, stream>>>(h16, edge_links, wmsgT, msg_b, agg, l);
    node_update<<<dim3(node_blocks, B), 256, 0, stream>>>(h16, agg, wupdT, upd_b, l);
  }

  kv_proj<<<dim3(node_blocks, B), 256, 0, stream>>>(h16, wkT, wvT, bk, bv, k16, v16);
  aq_kernel<<<1, 256, 0, stream>>>(current_ad, w_ad, b_ad, wq, bq, a_buf, q_buf);
  attn_kernel<<<B * HEADS, 256, 0, stream>>>(k16, v16, q_buf, ctx);
  g_kernel<<<1, 256, 0, stream>>>(a_buf, ctx, wo, bo, ln_g, ln_b, g_buf);
  logits_kernel<<<(B * N + 255) / 256, 256, 0, stream>>>(h16, g_buf, policy_w, policy_b,
                                                         mask, (float*)d_out);
}
